// EETransformerEncoder_89300960018629
// MI455X (gfx1250) — compile-verified, matrix operand order checked by reading, not run on hardware
//
#include <hip/hip_runtime.h>

typedef __attribute__((ext_vector_type(16))) __bf16 v16bf;
typedef __attribute__((ext_vector_type(8)))  float  v8f;

#define BM 128
#define BN 128
#define BKK 32

// LDS byte offset (addrspace(3) pointer value) of a generic pointer to shared.
__device__ __forceinline__ unsigned to_lds_off(const void* p) {
  return (unsigned)(uintptr_t)(const __attribute__((address_space(3))) char*)p;
}

// Async-DMA 32 bytes (one thread's staging slot) from global to LDS.
// ISA adds INST_OFFSET to both the LDS and global addresses, so offset:16
// covers the second 16B beat with the same operands. Tracked by ASYNCcnt.
__device__ __forceinline__ void async_copy32(unsigned ldsOff, const __bf16* g) {
  asm volatile("global_load_async_to_lds_b128 %0, %1, off\n\t"
               "global_load_async_to_lds_b128 %0, %1, off offset:16"
               :: "v"(ldsOff), "v"(g) : "memory");
}

__device__ __forceinline__ void wait_async0() {
  asm volatile("s_wait_asynccnt 0x0" ::: "memory");
}

// ---------------------------------------------------------------------------
// WMMA GEMM core, fast-path only: C(128x128 tile) += A[.,K] * B[.,K]^T
// (bf16 operands, K-contiguous rows, fp32 accumulate). Double-buffered LDS
// filled by GLOBAL_LOAD_ASYNC_TO_LDS_B128; one barrier per k-step; DMA for
// tile k+1 runs while tile k feeds the matrix pipes.
// REQUIREMENTS (guaranteed by callers via padding):
//   K % 32 == 0, lda/ldb % 8 == 0, all rows [bm0,bm0+128) / [bn0,bn0+128)
//   readable, 16B-aligned bases.
// 256 threads = 8 waves; each wave owns 32(M) x 64(N) = 2x4 WMMA accum tiles.
// ---------------------------------------------------------------------------
__device__ __forceinline__ void gemm_core(
    const __bf16* __restrict__ A, int lda,
    const __bf16* __restrict__ B, int ldb,
    int K, int bm0, int bn0,
    __bf16* lsA, __bf16* lsB,      // each: 2 buffers of BM*BKK
    v8f acc[2][4])
{
  const int tid   = threadIdx.x;
  const int lane  = tid & 31;
  const int wm    = (tid >> 5) & 3;   // wave M strip (0..3) -> 32 rows
  const int wn    = tid >> 7;         // wave N strip (0..1) -> 64 cols
  const int ldRow = tid >> 1;         // 0..127
  const int ldCh  = (tid & 1) << 4;   // 0 or 16

  const __bf16* aSrc = A + (long)(bm0 + ldRow) * lda + ldCh;
  const __bf16* bSrc = B + (long)(bn0 + ldRow) * ldb + ldCh;
  const unsigned aOff0 = to_lds_off(lsA + ldRow * BKK + ldCh);
  const unsigned bOff0 = to_lds_off(lsB + ldRow * BKK + ldCh);
  const unsigned bufStride = BM * BKK * 2;   // bytes per LDS buffer

  // kick off DMA for tile 0 into buffer 0
  async_copy32(aOff0, aSrc);
  async_copy32(bOff0, bSrc);

  const int kh = (lane >> 4) << 4;    // K half select per lane group
  const int r  = lane & 15;
  const int nk = K / BKK;

  for (int ki = 0; ki < nk; ++ki) {
    wait_async0();        // this wave's DMAs (current buffer) landed in LDS
    __syncthreads();      // all waves' tiles visible
    const int cur = ki & 1;
    // launch DMA for the next tile into the other buffer (overlaps WMMA)
    if (ki + 1 < nk) {
      const unsigned nb = (unsigned)((ki + 1) & 1) * bufStride;
      async_copy32(aOff0 + nb, aSrc + (ki + 1) * BKK);
      async_copy32(bOff0 + nb, bSrc + (ki + 1) * BKK);
    }
    const __bf16* la = lsA + cur * (BM * BKK);
    const __bf16* lb = lsB + cur * (BN * BKK);
    v16bf af[2], bfr[4];
    af[0] = *(const v16bf*)(la + (wm * 32 +      r) * BKK + kh);
    af[1] = *(const v16bf*)(la + (wm * 32 + 16 + r) * BKK + kh);
#pragma unroll
    for (int j = 0; j < 4; ++j)
      bfr[j] = *(const v16bf*)(lb + (wn * 64 + j * 16 + r) * BKK + kh);
#pragma unroll
    for (int i = 0; i < 2; ++i)
#pragma unroll
      for (int j = 0; j < 4; ++j)
        acc[i][j] = __builtin_amdgcn_wmma_f32_16x16x32_bf16(
            false, af[i], false, bfr[j], (short)0, acc[i][j], false, false);
  }
}

// ---------------------------------------------------------------------------
// Generic batched GEMM: C = A * B^T (+bias)(+residual)(ReLU?)(bf16 out?)
// Batch offsets decomposed as z = zb*HB + zh for head-sliced operands.
// M, N, K are padded compute dims; Ns bounds the epilogue stores (N-padding).
// flags: 1=ReLU, 2=store bf16
// ---------------------------------------------------------------------------
__global__ __launch_bounds__(256)
void wmma_gemm(const __bf16* __restrict__ A, const __bf16* __restrict__ B,
               void* __restrict__ C, const float* __restrict__ bias,
               const float* __restrict__ residual,
               int lda, int ldb, int ldc,
               int HB, long sAb, long sAh, long sBb, long sBh, long sCb, long sCh,
               int M, int N, int K, int Ns, int flags)
{
  __shared__ __attribute__((aligned(32))) __bf16 lsA[2 * BM * BKK];
  __shared__ __attribute__((aligned(32))) __bf16 lsB[2 * BN * BKK];

  const int z  = blockIdx.z;
  const int zb = z / HB, zh = z % HB;
  A += (long)zb * sAb + (long)zh * sAh;
  B += (long)zb * sBb + (long)zh * sBh;
  const long offC = (long)zb * sCb + (long)zh * sCh;

  const int bm0 = blockIdx.y * BM;
  const int bn0 = blockIdx.x * BN;

  v8f acc[2][4];
#pragma unroll
  for (int i = 0; i < 2; ++i)
#pragma unroll
    for (int j = 0; j < 4; ++j) acc[i][j] = (v8f)(0.0f);

  gemm_core(A, lda, B, ldb, K, bm0, bn0, lsA, lsB, acc);

  const int lane = threadIdx.x & 31;
  const int wm   = (threadIdx.x >> 5) & 3;
  const int wn   = threadIdx.x >> 7;
  const int hs   = lane >> 4;      // M offset 0/8 within 16x16 tile
  const int cn   = lane & 15;      // N within tile
#pragma unroll
  for (int i = 0; i < 2; ++i) {
#pragma unroll
    for (int j = 0; j < 4; ++j) {
      int n = bn0 + wn * 64 + j * 16 + cn;
      if (n >= Ns) continue;
      float bv = bias ? bias[n] : 0.f;
#pragma unroll
      for (int v = 0; v < 8; ++v) {
        int m = bm0 + wm * 32 + i * 16 + hs * 8 + v;
        if (m >= M) continue;
        float val = acc[i][j][v] + bv;
        if (residual) val += residual[offC + (long)m * ldc + n];
        if (flags & 1) val = fmaxf(val, 0.f);
        if (flags & 2) ((__bf16*)C)[offC + (long)m * ldc + n] = (__bf16)val;
        else           ((float*)C)[offC + (long)m * ldc + n] = val;
      }
    }
  }
}

// ---------------------------------------------------------------------------
// Attention scores: S[z,t,s] = (q_z[t,:]·k_z[s,:] + QP[z,t,t-s+511]) * 0.125
// masked to -FLT_MAX where mask[b,s]==0.  z = b*H + h.  T=512, DK=64.
// ---------------------------------------------------------------------------
__global__ __launch_bounds__(256)
void attn_scores(const __bf16* __restrict__ Q, const __bf16* __restrict__ Kb,
                 const float* __restrict__ QP, const int* __restrict__ masks,
                 float* __restrict__ S)
{
  __shared__ __attribute__((aligned(32))) __bf16 lsA[2 * BM * BKK];
  __shared__ __attribute__((aligned(32))) __bf16 lsB[2 * BN * BKK];

  const int z = blockIdx.z;
  const int b = z >> 2;          // H = 4
  const int h = z & 3;
  const __bf16* Az = Q  + (long)b * (512 * 256) + h * 64;
  const __bf16* Bz = Kb + (long)b * (512 * 256) + h * 64;
  const float*  QPz = QP + (long)z * 512 * 1024;
  float*        Sz  = S  + (long)z * 512 * 512;
  const int*    mb  = masks + b * 512;

  const int bm0 = blockIdx.y * BM;
  const int bn0 = blockIdx.x * BN;

  v8f acc[2][4];
#pragma unroll
  for (int i = 0; i < 2; ++i)
#pragma unroll
    for (int j = 0; j < 4; ++j) acc[i][j] = (v8f)(0.0f);

  gemm_core(Az, 256, Bz, 256, 64, bm0, bn0, lsA, lsB, acc);

  const int lane = threadIdx.x & 31;
  const int wm   = (threadIdx.x >> 5) & 3;
  const int wn   = threadIdx.x >> 7;
  const int hs   = lane >> 4;
  const int cn   = lane & 15;
#pragma unroll
  for (int i = 0; i < 2; ++i) {
#pragma unroll
    for (int j = 0; j < 4; ++j) {
      int n = bn0 + wn * 64 + j * 16 + cn;           // s
      int masked = (mb[n] == 0);
#pragma unroll
      for (int v = 0; v < 8; ++v) {
        int m = bm0 + wm * 32 + i * 16 + hs * 8 + v; // t
        float val = (acc[i][j][v] + QPz[(long)m * 1024 + (m - n + 511)]) * 0.125f;
        if (masked) val = -3.4028235e38f;
        Sz[(long)m * 512 + n] = val;
      }
    }
  }
}

// ---------------------------------------------------------------------------
// LayerNorm, one wave per row. flags: 1=ReLU after, 2=store bf16.
// ---------------------------------------------------------------------------
__global__ void layernorm_kernel(const float* __restrict__ in, void* __restrict__ out,
                                 const float* __restrict__ g, const float* __restrict__ bb,
                                 int rows, int D, int flags)
{
  int lane = threadIdx.x & 31;
  int wave = threadIdx.x >> 5;
  long row = (long)blockIdx.x * (blockDim.x >> 5) + wave;
  if (row >= rows) return;
  const float* x = in + row * D;
  float s = 0.f;
  for (int i = lane; i < D; i += 32) s += x[i];
  for (int o = 16; o; o >>= 1) s += __shfl_xor(s, o, 32);
  float mean = s / (float)D;
  float v = 0.f;
  for (int i = lane; i < D; i += 32) { float d = x[i] - mean; v += d * d; }
  for (int o = 16; o; o >>= 1) v += __shfl_xor(v, o, 32);
  float rstd = rsqrtf(v / (float)D + 1e-12f);
  for (int i = lane; i < D; i += 32) {
    float val = (x[i] - mean) * rstd * g[i] + bb[i];
    if (flags & 1) val = fmaxf(val, 0.f);
    if (flags & 2) ((__bf16*)out)[row * D + i] = (__bf16)val;
    else           ((float*)out)[row * D + i] = val;
  }
}

// ---------------------------------------------------------------------------
// Row softmax over s (T=512), one wave per (z,t) row; writes bf16 probs.
// ---------------------------------------------------------------------------
__global__ void softmax_kernel(const float* __restrict__ scores,
                               __bf16* __restrict__ attn,
                               const int* __restrict__ masks)
{
  int lane = threadIdx.x & 31;
  int wave = threadIdx.x >> 5;
  long row = (long)blockIdx.x * 8 + wave;   // [0, 32*512)
  int b = (int)((row >> 9) >> 2);           // z = row/512 ; b = z/H
  const float* src = scores + row * 512;
  float vals[16];
  float mx = -3.4028235e38f;
#pragma unroll
  for (int i = 0; i < 16; ++i) { vals[i] = src[lane + i * 32]; mx = fmaxf(mx, vals[i]); }
  for (int o = 16; o; o >>= 1) mx = fmaxf(mx, __shfl_xor(mx, o, 32));
  float sum = 0.f;
#pragma unroll
  for (int i = 0; i < 16; ++i) { vals[i] = __expf(vals[i] - mx); sum += vals[i]; }
  for (int o = 16; o; o >>= 1) sum += __shfl_xor(sum, o, 32);
  float inv = 1.f / sum;
#pragma unroll
  for (int i = 0; i < 16; ++i) {
    int sidx = lane + i * 32;
    float p = vals[i] * inv;
    if (masks[b * 512 + sidx] == 0) p = 0.f;
    attn[row * 512 + sidx] = (__bf16)p;
  }
}

// fp32 -> bf16 with output pitch; pad columns [cols, ldo) are written as 0
// so padded-K GEMMs read exact zeros (workspace is poisoned otherwise).
__global__ void cvt2d(const float* __restrict__ in, __bf16* __restrict__ out,
                      long n, int cols, int ldo)
{
  long i = (long)blockIdx.x * blockDim.x + threadIdx.x;
  if (i >= n) return;                 // n = rows * ldo
  long r = i / ldo; int c = (int)(i - r * ldo);
  out[i] = (c < cols) ? (__bf16)in[r * cols + c] : (__bf16)0.f;
}

// v[B,T,D] (bf16) -> vt[B,H,128,T] (bf16), rows d in [64,128) zero-padded so
// the attn@v GEMM runs a full in-bounds N=128 tile.
__global__ void transpose_v(const __bf16* __restrict__ v, __bf16* __restrict__ vt)
{
  long i = (long)blockIdx.x * blockDim.x + threadIdx.x;
  if (i >= (long)32 * 128 * 512) return;
  int t = (int)(i & 511); long r = i >> 9;
  int d = (int)(r & 127); r >>= 7;
  int h = (int)(r & 3);   int b = (int)(r >> 2);
  vt[i] = (d < 64) ? v[((long)(b * 512 + t)) * 256 + h * 64 + d] : (__bf16)0.f;
}

// ---------------------------------------------------------------------------
extern "C" void kernel_launch(void* const* d_in, const int* in_sizes, int n_in,
                              void* d_out, int out_size, void* d_ws, size_t ws_size,
                              hipStream_t stream)
{
  (void)in_sizes; (void)n_in; (void)out_size;
  const float* xs     = (const float*)d_in[0];
  const int*   masks  = (const int*)  d_in[1];
  const float* emb_w  = (const float*)d_in[2];
  const float* emb_b  = (const float*)d_in[3];
  const float* emb_g  = (const float*)d_in[4];
  const float* emb_be = (const float*)d_in[5];
  const float* pe_k   = (const float*)d_in[6];
  const float* Wq = (const float*)d_in[7];  const float* bq = (const float*)d_in[8];
  const float* Wk = (const float*)d_in[9];  const float* bk = (const float*)d_in[10];
  const float* Wv = (const float*)d_in[11]; const float* bv = (const float*)d_in[12];
  const float* Wo = (const float*)d_in[13]; const float* bo = (const float*)d_in[14];
  const float* ln1g = (const float*)d_in[15]; const float* ln1b = (const float*)d_in[16];
  const float* ln2g = (const float*)d_in[17]; const float* ln2b = (const float*)d_in[18];
  const float* lnkg = (const float*)d_in[19]; const float* lnkb = (const float*)d_in[20];
  const float* W1 = (const float*)d_in[21]; const float* b1 = (const float*)d_in[22];
  const float* W2 = (const float*)d_in[23]; const float* b2 = (const float*)d_in[24];
  const float* afg = (const float*)d_in[25]; const float* afb = (const float*)d_in[26];

  const int Bq = 8, T = 512, D = 256, H = 4, FF = 2048, L = 16, DK = 64, IDIM = 1799;
  const int MT  = Bq * T;          // 4096
  const int KP  = 1824;            // IDIM padded to a multiple of 32 (and 8)
  const int ZB  = Bq * H;          // 32 batched heads

  char* base = (char*)d_ws;
  size_t off = 0;
  auto alloc = [&](size_t bytes) -> void* {
    void* p = base + off;
    off += (bytes + 255) & ~(size_t)255;
    return p;
  };
  __bf16* xsb  = (__bf16*)alloc((size_t)MT * KP * 2);
  __bf16* wbuf = (__bf16*)alloc((size_t)FF * D * 2);        // >= 256*1824 too
  float*  x    = (float*) alloc((size_t)MT * D * 4);        // residual stream
  __bf16* ybf  = (__bf16*)alloc((size_t)MT * D * 2);
  __bf16* pkb  = (__bf16*)alloc((size_t)2000 * DK * 2);
  __bf16* qb   = (__bf16*)alloc((size_t)MT * D * 2);
  __bf16* kb_  = (__bf16*)alloc((size_t)MT * D * 2);
  __bf16* vb   = (__bf16*)alloc((size_t)MT * D * 2);
  __bf16* vtb  = (__bf16*)alloc((size_t)ZB * 128 * T * 2);  // padded rows
  __bf16* ob   = (__bf16*)alloc((size_t)MT * D * 2);
  float*  QP   = (float*) alloc((size_t)ZB * T * 1024 * 4);
  float*  sc   = (float*) alloc((size_t)ZB * T * T * 4);
  __bf16* attn = (__bf16*)alloc((size_t)ZB * T * T * 2);
  __bf16* h1   = (__bf16*)alloc((size_t)MT * FF * 2);
  if (off > ws_size) return;     // scratch undersized: bail safely

  auto cdiv = [](long a, long b) { return (int)((a + b - 1) / b); };
  auto cvt = [&](const float* in, __bf16* out, long rows, int cols, int ldo) {
    long n = rows * ldo;
    cvt2d<<<cdiv(n, 256), 256, 0, stream>>>(in, out, n, cols, ldo);
  };
  auto ln = [&](const float* in, void* out, const float* g, const float* bb,
                int rows, int Dd, int flags) {
    layernorm_kernel<<<cdiv(rows, 8), 256, 0, stream>>>(in, out, g, bb, rows, Dd, flags);
  };
  auto gemm = [&](const __bf16* A, const __bf16* Bp, void* C,
                  const float* bias, const float* resid,
                  int lda, int ldb, int ldc,
                  int HB, long sAb, long sAh, long sBb, long sBh, long sCb, long sCh,
                  int M, int N, int K, int Ns, int flags, int nb) {
    dim3 g(cdiv(N, BN), cdiv(M, BM), nb);
    wmma_gemm<<<g, 256, 0, stream>>>(A, Bp, C, bias, resid, lda, ldb, ldc,
                                     HB, sAb, sAh, sBb, sBh, sCb, sCh,
                                     M, N, K, Ns, flags);
  };

  // ---- embed: Linear(IDIM->D) -> LN -> ReLU --------------------------------
  cvt(xs, xsb, MT, IDIM, KP);            // zero-pads cols [1799,1824)
  cvt(emb_w, wbuf, D, IDIM, KP);
  float* tmpf = QP;   // reuse QP slab as fp32 temp
  gemm(xsb, wbuf, tmpf, emb_b, nullptr, KP, KP, D,
       1, 0, 0, 0, 0, 0, 0, MT, D, KP, D, 0, 1);
  ln(tmpf, x, emb_g, emb_be, MT, D, 1 /*ReLU, fp32 out*/);

  // ---- layers --------------------------------------------------------------
  for (int l = 0; l < L; ++l) {
    const float* Wq_l = Wq + (long)l * D * D;  const float* bq_l = bq + l * D;
    const float* Wk_l = Wk + (long)l * D * D;  const float* bk_l = bk + l * D;
    const float* Wv_l = Wv + (long)l * D * D;  const float* bv_l = bv + l * D;
    const float* Wo_l = Wo + (long)l * D * D;  const float* bo_l = bo + l * D;
    const float* W1_l = W1 + (long)l * FF * D; const float* b1_l = b1 + l * FF;
    const float* W2_l = W2 + (long)l * D * FF; const float* b2_l = b2 + l * D;

    // norm_k of rel-pos table (LN over DK), then ln1 of stream
    ln(pe_k, pkb, lnkg + l * DK, lnkb + l * DK, 2000, DK, 2 /*bf16*/);
    ln(x, ybf, ln1g + l * D, ln1b + l * D, MT, D, 2 /*bf16*/);

    // q, k, v projections (bf16 out)
    cvt(Wq_l, wbuf, D, D, D);
    gemm(ybf, wbuf, qb, bq_l, nullptr, D, D, D, 1,0,0,0,0,0,0, MT, D, D, D, 2, 1);
    cvt(Wk_l, wbuf, D, D, D);
    gemm(ybf, wbuf, kb_, bk_l, nullptr, D, D, D, 1,0,0,0,0,0,0, MT, D, D, D, 2, 1);
    cvt(Wv_l, wbuf, D, D, D);
    gemm(ybf, wbuf, vb, bv_l, nullptr, D, D, D, 1,0,0,0,0,0,0, MT, D, D, D, 2, 1);
    transpose_v<<<cdiv((long)ZB * 128 * T, 256), 256, 0, stream>>>(vb, vtb);

    // QP[z,t,r] = q_z[t,:] · pk[489+r,:]   (r = t-s+511, rows 489..1511)
    gemm(qb, pkb + 489 * DK, QP, nullptr, nullptr,
         D, DK, 1024,
         H, (long)T * D, DK, 0, 0, (long)H * T * 1024, (long)T * 1024,
         T, 1024, DK, 1024, 0, ZB);

    // scores = (q·k^T + gather(QP)) * 1/sqrt(DK), masked
    attn_scores<<<dim3(4, 4, ZB), 256, 0, stream>>>(qb, kb_, QP, masks, sc);

    // softmax rows -> bf16 probabilities
    softmax_kernel<<<(ZB * T) / 8, 256, 0, stream>>>(sc, attn, masks);

    // o = attn @ v : N padded to 128 (zero rows), store bound Ns = 64
    gemm(attn, vtb, ob, nullptr, nullptr,
         T, T, D,
         H, (long)H * T * T, (long)T * T, (long)H * 128 * T, (long)128 * T,
         (long)T * D, DK,
         T, 128, T, DK, 2, ZB);

    // x = x + o @ Wo^T + bo
    cvt(Wo_l, wbuf, D, D, D);
    gemm(ob, wbuf, x, bo_l, x, D, D, D, 1,0,0,0,0,0,0, MT, D, D, D, 0, 1);

    // FF
    ln(x, ybf, ln2g + l * D, ln2b + l * D, MT, D, 2 /*bf16*/);
    cvt(W1_l, wbuf, FF, D, D);
    gemm(ybf, wbuf, h1, b1_l, nullptr, D, D, FF, 1,0,0,0,0,0,0,
         MT, FF, D, FF, 3 /*ReLU|bf16*/, 1);
    cvt(W2_l, wbuf, D, FF, FF);
    gemm(h1, wbuf, x, b2_l, x, FF, FF, D, 1,0,0,0,0,0,0, MT, D, FF, D, 0, 1);
  }

  // ---- final LN -> d_out (fp32) -------------------------------------------
  ln(x, (float*)d_out, afg, afb, MT, D, 0);
}
